// PM25_GNN_MLP_87299505259220
// MI455X (gfx1250) — matrix operand order repaired and checked by reading, not compiled, not run on hardware
//
#include <hip/hip_runtime.h>
#include <hip/hip_bf16.h>

typedef __attribute__((ext_vector_type(16))) _Float16 v16h;
typedef __attribute__((ext_vector_type(8)))  _Float16 v8h;
typedef __attribute__((ext_vector_type(8)))  float    v8f;

// Problem constants
#define B_    16
#define HIST_ 8
#define PRED_ 12
#define NN    2000
#define EE    64000
#define FEAT_ 13
#define IND_  14      // IN_DIM = FEAT+1
#define HID_  64
#define GOUT  13
#define EH_   32
#define EOUT_ 30

__device__ __forceinline__ float sigmoidf(float v) {
  // 1/(1+exp(-v)) with HW reciprocal: v_mul+v_exp+v_add+v_rcp
  return __builtin_amdgcn_rcpf(1.0f + __expf(-v));
}

__device__ __forceinline__ v8f wmma_f16(v16h a, v16h b, v8f c) {
  return __builtin_amdgcn_wmma_f32_16x16x32_f16(false, a, false, b, (short)0, c,
                                                false, false);
}

// A-fragment (16x32 f16, wave32): lane L -> row m=L&15, half=L>>4
// elements 0..7  -> K = half*16 + i       (contiguous 16B)
// elements 8..15 -> K = half*16 + 8 + i   (contiguous 16B)
__device__ __forceinline__ v16h load_a_frag(const _Float16* tile, int lane,
                                            int stride, int kofs) {
  const int m = lane & 15;
  const int half = lane >> 4;
  const _Float16* p = tile + m * stride + kofs + half * 16;
  v8h lo = *(const v8h*)(p);
  v8h hi = *(const v8h*)(p + 8);
  return __builtin_shufflevector(lo, hi, 0, 1, 2, 3, 4, 5, 6, 7, 8, 9, 10, 11,
                                 12, 13, 14, 15);
}

// B-fragment from pre-swizzled weights: [kg][ntile][lane][16] halves contiguous
__device__ __forceinline__ v16h load_b_swz(const _Float16* Wsw, int lane,
                                           int NT, int nt, int kg) {
  return *(const v16h*)(Wsw + ((((kg * NT + nt) << 5) + lane) << 4));
}

// ------------------------------------------------------------------ setup

__global__ void stats_kernel(const float* __restrict__ eattr,
                             float* __restrict__ stats) {
  __shared__ float red[4][256];
  float s0 = 0, s1 = 0, q0 = 0, q1 = 0;
  for (int e = threadIdx.x; e < EE; e += 256) {
    float d = eattr[2 * e + 0];
    float w = eattr[2 * e + 1];
    s0 += d; q0 += d * d;
    s1 += w; q1 += w * w;
  }
  red[0][threadIdx.x] = s0; red[1][threadIdx.x] = q0;
  red[2][threadIdx.x] = s1; red[3][threadIdx.x] = q1;
  __syncthreads();
  for (int off = 128; off > 0; off >>= 1) {
    if ((int)threadIdx.x < off)
      for (int j = 0; j < 4; ++j)
        red[j][threadIdx.x] += red[j][threadIdx.x + off];
    __syncthreads();
  }
  if (threadIdx.x == 0) {
    const float n = (float)EE;
    float m0 = red[0][0] / n, m1 = red[2][0] / n;
    float v0 = (red[1][0] - n * m0 * m0) / (n - 1.0f);
    float v1 = (red[3][0] - n * m1 * m1) / (n - 1.0f);
    stats[0] = m0;
    stats[1] = m1;
    stats[2] = rsqrtf(fmaxf(v0, 1e-20f));
    stats[3] = rsqrtf(fmaxf(v1, 1e-20f));
  }
}

// Write weight (K x N, row-major f32) as f16 fragments:
// dst[((kg*NT + nt)*32 + lane)*16 + i] = W[kg*32 + (lane>>4)*16 + i][nt*16 + (lane&15)]
__device__ void swizzle_weight(const float* __restrict__ W,
                               _Float16* __restrict__ dst, int K, int N, int KG,
                               int NT, int ntShift, int tid, int nthreads) {
  const int total = KG * NT * 512;
  for (int idx = tid; idx < total; idx += nthreads) {
    const int i = idx & 15;
    const int lane = (idx >> 4) & 31;
    const int nt = (idx >> 9) & (NT - 1);
    const int kg = idx >> (9 + ntShift);
    const int k = kg * 32 + ((lane >> 4) << 4) + i;
    const int n = (nt << 4) + (lane & 15);
    dst[idx] = (k < K && n < N) ? (_Float16)W[k * N + n] : (_Float16)0.0f;
  }
}

__global__ void convert_weights_kernel(
    const float* __restrict__ We1, const float* __restrict__ We2,
    const float* __restrict__ Wn,  const float* __restrict__ W1,
    const float* __restrict__ W2,  const float* __restrict__ W3,
    _Float16* __restrict__ We1s, _Float16* __restrict__ We2s,
    _Float16* __restrict__ Wns,  _Float16* __restrict__ W1s,
    _Float16* __restrict__ W2s,  _Float16* __restrict__ W3s) {
  const int tid = threadIdx.x;
  swizzle_weight(We1, We1s, 31, 32, 1, 2, 1, tid, 256);
  swizzle_weight(We2, We2s, 32, 30, 1, 2, 1, tid, 256);
  swizzle_weight(Wn,  Wns,  30, 13, 1, 1, 0, tid, 256);
  swizzle_weight(W1,  W1s,  27, 64, 1, 4, 2, tid, 256);
  swizzle_weight(W2,  W2s,  64, 64, 2, 4, 2, tid, 256);
  swizzle_weight(W3,  W3s,  64, 64, 2, 4, 2, tid, 256);
}

__global__ void init_xn_kernel(const float* __restrict__ pm25,
                               float* __restrict__ xn) {
  int i = blockIdx.x * blockDim.x + threadIdx.x;
  if (i >= B_ * NN) return;
  int b = i / NN, n = i % NN;
  xn[i] = pm25[((long long)b * HIST_ + (HIST_ - 1)) * NN + n];
}

__global__ void build_x_kernel(const float* __restrict__ xn,
                               const float* __restrict__ feature,
                               float* __restrict__ x, int t) {
  int i = blockIdx.x * blockDim.x + threadIdx.x;
  if (i >= B_ * NN) return;
  int b = i / NN, n = i % NN;
  float* xr = x + (long long)i * IND_;
  xr[0] = xn[i];
  const float* fr =
      feature + (((long long)b * (HIST_ + PRED_) + (HIST_ + t)) * NN + n) * FEAT_;
#pragma unroll
  for (int f = 0; f < FEAT_; ++f) xr[1 + f] = fr[f];
}

__global__ void zero_kernel(float* __restrict__ p, int n) {
  int i = blockIdx.x * blockDim.x + threadIdx.x;
  if (i < n) p[i] = 0.0f;
}

// ------------------------------------------------------------------ edge MLP
// 4 waves/block; each wave owns 16 edge rows of one batch b.
// e = sigmoid(sigmoid([ns|nt|an|ew] @ We1 + be1) @ We2 + be2)
// agg[b,tgt] += e ; agg[b,src] -= e
__global__ __launch_bounds__(128) void edge_kernel(
    const float* __restrict__ x, const int* __restrict__ eidx,
    const float* __restrict__ eattr, const float* __restrict__ stats,
    const float* __restrict__ wind_mean, const float* __restrict__ wind_std,
    const _Float16* __restrict__ We1s, const float* __restrict__ be1,
    const _Float16* __restrict__ We2s, const float* __restrict__ be2,
    float* __restrict__ agg) {
  __shared__ __align__(32) _Float16 sW1[1024];   // swizzled (1 kg, 2 nt)
  __shared__ __align__(32) _Float16 sW2[1024];
  __shared__ __align__(32) _Float16 sIn[4][16 * 32];
  __shared__ __align__(32) _Float16 sMid[4][16 * 32];
  __shared__ float sOut[4][16 * 32];

  const int tid = threadIdx.x;
  for (int i = tid; i < 1024; i += 128) {
    sW1[i] = We1s[i];
    sW2[i] = We2s[i];
  }
  __syncthreads();

  const int wave = tid >> 5;
  const int lane = tid & 31;
  const long long rowBase = ((long long)blockIdx.x * 4 + wave) * 16;
  const int b = (int)(rowBase / EE);
  const int e0 = (int)(rowBase % EE);
  const int m = lane & 15;
  const int half = lane >> 4;

  // ---- stage 16 rows x 32 cols of edge-MLP input as f16 (vector stores)
  {
    const int e = e0 + m;
    const int s = eidx[e];
    const int tg = eidx[EE + e];
    const float* nsrow = x + ((long long)(b * NN + s)) * IND_;
    const float* ntrow = x + ((long long)(b * NN + tg)) * IND_;
    v8h lo, hi;
    if (half == 0) {
#pragma unroll
      for (int k = 0; k < 8; ++k) lo[k] = (_Float16)nsrow[k];
#pragma unroll
      for (int k = 0; k < 6; ++k) hi[k] = (_Float16)nsrow[8 + k];
      hi[6] = (_Float16)ntrow[0];
      hi[7] = (_Float16)ntrow[1];
    } else {
#pragma unroll
      for (int k = 0; k < 8; ++k) lo[k] = (_Float16)ntrow[2 + k];
#pragma unroll
      for (int k = 0; k < 4; ++k) hi[k] = (_Float16)ntrow[10 + k];
      float dist = eattr[e * 2 + 0];
      float direc = eattr[e * 2 + 1];
      hi[4] = (_Float16)((dist - stats[0]) * stats[2]);
      hi[5] = (_Float16)((direc - stats[1]) * stats[3]);
      float speed = nsrow[12] * wind_std[0] + wind_mean[0];
      float wdir = nsrow[13] * wind_std[1] + wind_mean[1];
      float theta = fabsf(direc - wdir);
      float ew = 3.0f * speed * __cosf(theta) / dist;
      hi[6] = (_Float16)(ew > 0.0f ? ew : 0.0f);
      hi[7] = (_Float16)0.0f;
    }
    _Float16* dst = &sIn[wave][m * 32 + half * 16];
    *(v8h*)(dst) = lo;
    *(v8h*)(dst + 8) = hi;
  }
  __syncthreads();

  // ---- layer 1: (16x31pad32) @ (32x32) via 2 WMMAs
  {
    v16h a = load_a_frag(sIn[wave], lane, 32, 0);
    const int n = lane & 15;
#pragma unroll
    for (int tile = 0; tile < 2; ++tile) {
      v8f c = {};
      c = wmma_f16(a, load_b_swz(sW1, lane, 2, tile, 0), c);
      const int nc = n + tile * 16;
#pragma unroll
      for (int v = 0; v < 8; ++v) {
        int mm = v + 8 * half;
        sMid[wave][mm * 32 + nc] = (_Float16)sigmoidf(c[v] + be1[nc]);
      }
    }
  }
  __syncthreads();

  // ---- layer 2: (16x32) @ (32x30pad32) via 2 WMMAs
  {
    v16h a = load_a_frag(sMid[wave], lane, 32, 0);
    const int n = lane & 15;
#pragma unroll
    for (int tile = 0; tile < 2; ++tile) {
      v8f c = {};
      c = wmma_f16(a, load_b_swz(sW2, lane, 2, tile, 0), c);
      const int nc = n + tile * 16;
      const float bias = (nc < EOUT_) ? be2[nc] : 0.0f;
#pragma unroll
      for (int v = 0; v < 8; ++v) {
        int mm = v + 8 * half;
        sOut[wave][mm * 32 + nc] = sigmoidf(c[v] + bias);
      }
    }
  }
  __syncthreads();

  // ---- scatter: agg[b,tgt]+=e ; agg[b,src]-=e
  for (int idx = lane; idx < 16 * EOUT_; idx += 32) {
    int mm = idx / EOUT_;
    int nc = idx % EOUT_;
    float v = sOut[wave][mm * 32 + nc];
    int e = e0 + mm;
    int s = eidx[e];
    int tg = eidx[EE + e];
    atomicAdd(&agg[((long long)(b * NN + tg)) * EOUT_ + nc], v);
    atomicAdd(&agg[((long long)(b * NN + s)) * EOUT_ + nc], -v);
  }
}

// ------------------------------------------------------------------ node MLP
// g = sigmoid(agg@Wn+bn); h=[g|x]; 3x sigmoid MLP; out = h3@Wo+bo
__global__ __launch_bounds__(128) void node_kernel(
    const float* __restrict__ x, const float* __restrict__ agg,
    const _Float16* __restrict__ Wns, const float* __restrict__ bn,
    const _Float16* __restrict__ W1s, const float* __restrict__ b1,
    const _Float16* __restrict__ W2s, const float* __restrict__ b2,
    const _Float16* __restrict__ W3s, const float* __restrict__ b3,
    const float* __restrict__ Wo, const float* __restrict__ bo,
    float* __restrict__ out, float* __restrict__ xn_next, int t) {
  __shared__ __align__(32) _Float16 sWn[512];    // 1 kg, 1 nt
  __shared__ __align__(32) _Float16 sW1[2048];   // 1 kg, 4 nt
  __shared__ __align__(32) _Float16 sW2[4096];   // 2 kg, 4 nt
  __shared__ __align__(32) _Float16 sW3[4096];
  __shared__ __align__(32) _Float16 sBufA[4][16 * 64];
  __shared__ __align__(32) _Float16 sBufB[4][16 * 64];
  __shared__ float sBufF[4][16 * 64];

  const int tid = threadIdx.x;
  for (int i = tid; i < 512; i += 128) sWn[i] = Wns[i];
  for (int i = tid; i < 2048; i += 128) sW1[i] = W1s[i];
  for (int i = tid; i < 4096; i += 128) {
    sW2[i] = W2s[i];
    sW3[i] = W3s[i];
  }
  __syncthreads();

  const int wave = tid >> 5;
  const int lane = tid & 31;
  const int r0 = (blockIdx.x * 4 + wave) * 16;
  const int b = r0 / NN;
  const int n0 = r0 % NN;
  const int m = lane & 15;
  const int half = lane >> 4;

  _Float16* bufA = sBufA[wave];
  _Float16* bufB = sBufB[wave];
  float* bufF = sBufF[wave];

  // ---- stage agg rows (16 x 30pad32) as f16 into bufA (vector stores)
  {
    const float* arow = agg + ((long long)(b * NN + n0 + m)) * EOUT_;
    v8h lo, hi;
#pragma unroll
    for (int c = 0; c < 8; ++c) {
      int K = half * 16 + c;
      lo[c] = (K < EOUT_) ? (_Float16)arow[K] : (_Float16)0.0f;
    }
#pragma unroll
    for (int c = 0; c < 8; ++c) {
      int K = half * 16 + 8 + c;
      hi[c] = (K < EOUT_) ? (_Float16)arow[K] : (_Float16)0.0f;
    }
    _Float16* dst = bufA + m * 32 + half * 16;
    *(v8h*)(dst) = lo;
    *(v8h*)(dst + 8) = hi;
  }

  // ---- x part of hin FIRST (full-row vector store; g cols overwritten later)
  {
    const float* xrow = x + ((long long)(b * NN + n0 + m)) * IND_;
    v8h lo, hi;
#pragma unroll
    for (int c = 0; c < 8; ++c) {
      int gc = half * 16 + c;
      lo[c] = (gc >= GOUT && gc < GOUT + IND_) ? (_Float16)xrow[gc - GOUT]
                                               : (_Float16)0.0f;
    }
#pragma unroll
    for (int c = 0; c < 8; ++c) {
      int gc = half * 16 + 8 + c;
      hi[c] = (gc >= GOUT && gc < GOUT + IND_) ? (_Float16)xrow[gc - GOUT]
                                               : (_Float16)0.0f;
    }
    _Float16* dst = bufB + m * 32 + half * 16;
    *(v8h*)(dst) = lo;
    *(v8h*)(dst + 8) = hi;
  }
  __syncthreads();

  // ---- g = sigmoid(agg @ Wn + bn) -> cols 0..12 of hin (bufB)
  {
    v16h a = load_a_frag(bufA, lane, 32, 0);
    v8f c = {};
    c = wmma_f16(a, load_b_swz(sWn, lane, 1, 0, 0), c);
    const int n = lane & 15;
    if (n < GOUT) {
#pragma unroll
      for (int v = 0; v < 8; ++v) {
        int mm = v + 8 * half;
        bufB[mm * 32 + n] = (_Float16)sigmoidf(c[v] + bn[n]);
      }
    }
  }
  __syncthreads();

  // ---- layer 1: hin(16x27pad32) @ W1(32x64) -> bufA (f16, stride 64)
  {
    v16h a = load_a_frag(bufB, lane, 32, 0);
#pragma unroll
    for (int tile = 0; tile < 4; ++tile) {
      v8f c = {};
      c = wmma_f16(a, load_b_swz(sW1, lane, 4, tile, 0), c);
      const int n = (lane & 15) + tile * 16;
#pragma unroll
      for (int v = 0; v < 8; ++v) {
        int mm = v + 8 * half;
        bufA[mm * 64 + n] = (_Float16)sigmoidf(c[v] + b1[n]);
      }
    }
  }
  __syncthreads();

  // ---- layer 2: h1(16x64) @ W2(64x64) -> bufB (f16, stride 64)
  {
    v16h aLo = load_a_frag(bufA, lane, 64, 0);
    v16h aHi = load_a_frag(bufA, lane, 64, 32);
#pragma unroll
    for (int tile = 0; tile < 4; ++tile) {
      v8f c = {};
      c = wmma_f16(aLo, load_b_swz(sW2, lane, 4, tile, 0), c);
      c = wmma_f16(aHi, load_b_swz(sW2, lane, 4, tile, 1), c);
      const int n = (lane & 15) + tile * 16;
#pragma unroll
      for (int v = 0; v < 8; ++v) {
        int mm = v + 8 * half;
        bufB[mm * 64 + n] = (_Float16)sigmoidf(c[v] + b2[n]);
      }
    }
  }
  __syncthreads();

  // ---- layer 3: h2(16x64) @ W3(64x64) -> bufF (f32)
  {
    v16h aLo = load_a_frag(bufB, lane, 64, 0);
    v16h aHi = load_a_frag(bufB, lane, 64, 32);
#pragma unroll
    for (int tile = 0; tile < 4; ++tile) {
      v8f c = {};
      c = wmma_f16(aLo, load_b_swz(sW3, lane, 4, tile, 0), c);
      c = wmma_f16(aHi, load_b_swz(sW3, lane, 4, tile, 1), c);
      const int n = (lane & 15) + tile * 16;
#pragma unroll
      for (int v = 0; v < 8; ++v) {
        int mm = v + 8 * half;
        bufF[mm * 64 + n] = sigmoidf(c[v] + b3[n]);
      }
    }
  }
  __syncthreads();

  // ---- output: 64->1 dot per row (lanes 0..15)
  if (lane < 16) {
    float s = bo[0];
#pragma unroll 8
    for (int k = 0; k < 64; ++k) s += bufF[lane * 64 + k] * Wo[k];
    int n = n0 + lane;
    out[((long long)(b * PRED_ + t)) * NN + n] = s;
    xn_next[b * NN + n] = s;
  }
}

// ------------------------------------------------------------------ host

extern "C" void kernel_launch(void* const* d_in, const int* in_sizes, int n_in,
                              void* d_out, int out_size, void* d_ws,
                              size_t ws_size, hipStream_t stream) {
  (void)in_sizes; (void)n_in; (void)out_size; (void)ws_size;
  const float* pm25      = (const float*)d_in[0];
  const float* feature   = (const float*)d_in[1];
  const float* edge_attr = (const float*)d_in[2];
  const float* wind_mean = (const float*)d_in[3];
  const float* wind_std  = (const float*)d_in[4];
  const float* We1 = (const float*)d_in[5];
  const float* be1 = (const float*)d_in[6];
  const float* We2 = (const float*)d_in[7];
  const float* be2 = (const float*)d_in[8];
  const float* Wn  = (const float*)d_in[9];
  const float* bn  = (const float*)d_in[10];
  const float* W1  = (const float*)d_in[11];
  const float* b1  = (const float*)d_in[12];
  const float* W2  = (const float*)d_in[13];
  const float* b2  = (const float*)d_in[14];
  const float* W3  = (const float*)d_in[15];
  const float* b3  = (const float*)d_in[16];
  const float* Wo  = (const float*)d_in[17];
  const float* bo  = (const float*)d_in[18];
  const int*   eidx = (const int*)d_in[19];
  float* out = (float*)d_out;

  char* ws = (char*)d_ws;
  size_t off = 0;
  auto alloc = [&](size_t bytes) -> void* {
    void* p = ws + off;
    off += bytes;
    off = (off + 255) & ~(size_t)255;
    return p;
  };
  float* stats = (float*)alloc(4 * sizeof(float));
  float* xn    = (float*)alloc((size_t)B_ * NN * sizeof(float));
  float* x     = (float*)alloc((size_t)B_ * NN * IND_ * sizeof(float));
  float* agg   = (float*)alloc((size_t)B_ * NN * EOUT_ * sizeof(float));
  _Float16* We1s = (_Float16*)alloc(1024 * sizeof(_Float16));
  _Float16* We2s = (_Float16*)alloc(1024 * sizeof(_Float16));
  _Float16* Wns  = (_Float16*)alloc(512 * sizeof(_Float16));
  _Float16* W1s  = (_Float16*)alloc(2048 * sizeof(_Float16));
  _Float16* W2s  = (_Float16*)alloc(4096 * sizeof(_Float16));
  _Float16* W3s  = (_Float16*)alloc(4096 * sizeof(_Float16));

  stats_kernel<<<1, 256, 0, stream>>>(edge_attr, stats);
  convert_weights_kernel<<<1, 256, 0, stream>>>(We1, We2, Wn, W1, W2, W3, We1s,
                                                We2s, Wns, W1s, W2s, W3s);
  init_xn_kernel<<<(B_ * NN + 255) / 256, 256, 0, stream>>>(pm25, xn);

  const int edgeBlocks = (B_ * EE) / 64;   // 16000
  const int nodeBlocks = (B_ * NN) / 64;   // 500
  const int aggElems = B_ * NN * EOUT_;

  for (int t = 0; t < PRED_; ++t) {
    build_x_kernel<<<(B_ * NN + 255) / 256, 256, 0, stream>>>(xn, feature, x, t);
    zero_kernel<<<(aggElems + 255) / 256, 256, 0, stream>>>(agg, aggElems);
    edge_kernel<<<edgeBlocks, 128, 0, stream>>>(x, eidx, edge_attr, stats,
                                                wind_mean, wind_std, We1s, be1,
                                                We2s, be2, agg);
    node_kernel<<<nodeBlocks, 128, 0, stream>>>(x, agg, Wns, bn, W1s, b1, W2s,
                                                b2, W3s, b3, Wo, bo, out, xn, t);
  }
}
